// GlobalSLCLayer_33285996544384
// MI455X (gfx1250) — matrix operand order repaired and checked by reading, not compiled, not run on hardware
//
#include <hip/hip_runtime.h>

typedef __attribute__((ext_vector_type(16))) __bf16 v16bf;
typedef __attribute__((ext_vector_type(8)))  __bf16 v8bf;
typedef __attribute__((ext_vector_type(8)))  float  v8f;

#define WPB 8   // waves per block (blockDim = 256, wave32)

// ---- helpers ---------------------------------------------------------------

__device__ __forceinline__ v16bf load_afrag(const __bf16* __restrict__ row, int kb, int half) {
    // A 16x32 bf16 fragment: elems 0..7 -> K = kb + 16*half + 0..7
    //                        elems 8..15 -> K = kb + 16*half + 8 + 0..7
    v8bf lo = *(const v8bf*)(row + kb + 16 * half);
    v8bf hi = *(const v8bf*)(row + kb + 16 * half + 8);
    return __builtin_shufflevector(lo, hi, 0,1,2,3,4,5,6,7,8,9,10,11,12,13,14,15);
}

__device__ __forceinline__ v8f wmma_bf16(v16bf a, v16bf b, v8f c) {
    return __builtin_amdgcn_wmma_f32_16x16x32_bf16(false, a, false, b, (short)0, c, false, false);
}

// ---- conversion / transpose kernels ---------------------------------------

__global__ void cvt_f32_bf16(const float* __restrict__ in, __bf16* __restrict__ outp, long long n) {
    long long i = (long long)blockIdx.x * blockDim.x + threadIdx.x;
    if (i < n) outp[i] = (__bf16)in[i];
}

// outT[b][c][r] = (bf16) in[b][r][c]   (rows x cols per batch, batch = blockIdx.y)
__global__ void transpose_cvt(const float* __restrict__ in, __bf16* __restrict__ outT,
                              int rows, int cols) {
    long long batchOff = (long long)blockIdx.y * rows * cols;
    int idx = blockIdx.x * blockDim.x + threadIdx.x;
    if (idx < rows * cols) {
        int r = idx / cols, c = idx % cols;
        outT[batchOff + (long long)c * rows + r] = (__bf16)in[batchOff + idx];
    }
}

// ---- generic WMMA GEMM: C[M x 64] = alpha * (A[M x K] * B[K x 64]) - D -----
// A row-major bf16 (lda), B supplied TRANSPOSED row-major bf16 Bt[64 x K] (ldb).
// Outputs (each optional): Cf  f32 row-major [M x 64]
//                          Crm bf16 row-major [M x 64]
//                          Ct  bf16 transposed [64 x M]
__global__ __launch_bounds__(256) void gemm_wmma(
    const __bf16* __restrict__ A,  long long aBatch, int lda,
    const __bf16* __restrict__ Bt, long long btBatch, int ldb,
    const float*  __restrict__ D,  long long dBatch,
    float*  __restrict__ Cf, __bf16* __restrict__ Crm, long long cBatch,
    __bf16* __restrict__ Ct, long long ctBatch,
    int M, int K, float alpha)
{
    const int lane = threadIdx.x & 31;
    const int wave = threadIdx.x >> 5;
    const int b    = blockIdx.y;
    const int m0   = (blockIdx.x * WPB + wave) * 16;
    if (m0 >= M) return;                       // wave-uniform exit (EXEC stays all-1s)

    const int mrow = lane & 15;
    const int half = lane >> 4;
    const __bf16* aRow  = A  + (long long)b * aBatch + (long long)(m0 + mrow) * lda;
    const __bf16* bBase = Bt + (long long)b * btBatch;

    v8f acc[4] = {};
    for (int kb = 0; kb < K; kb += 32) {
        v16bf af = load_afrag(aRow, kb, half);
        #pragma unroll
        for (int nt = 0; nt < 4; ++nt) {
            const __bf16* bp = bBase + (long long)(nt * 16 + mrow) * ldb + kb + 16 * half;
            v16bf bf_ = *(const v16bf*)bp;
            acc[nt] = wmma_bf16(af, bf_, acc[nt]);
        }
    }

    const int mbase = m0 + 8 * half;
    #pragma unroll
    for (int nt = 0; nt < 4; ++nt) {
        const int n = nt * 16 + mrow;
        v8bf ctv;
        #pragma unroll
        for (int r = 0; r < 8; ++r) {
            float v = alpha * acc[nt][r];
            if (D)   v -= D[(long long)b * dBatch + (long long)(mbase + r) * 64 + n];
            if (Cf)  Cf [(long long)b * cBatch + (long long)(mbase + r) * 64 + n] = v;
            if (Crm) Crm[(long long)b * cBatch + (long long)(mbase + r) * 64 + n] = (__bf16)v;
            ctv[r] = (__bf16)v;
        }
        if (Ct) *(v8bf*)(Ct + (long long)b * ctBatch + (long long)n * M + mbase) = ctv;
    }
}

// ---- fused output: relu(sum_k Yk*Ths[k]) + relu(sum_k Zk*Thd[k]) -----------
// Yk/Zk row-major bf16 [2048 x 64] per batch; ThsT/ThdT bf16 [4][64 x 64] transposed.
__global__ __launch_bounds__(256) void final_wmma(
    const __bf16* __restrict__ Y0, const __bf16* __restrict__ Y1,
    const __bf16* __restrict__ Y2, const __bf16* __restrict__ Y3,
    const __bf16* __restrict__ Z1, const __bf16* __restrict__ Z2,
    const __bf16* __restrict__ Z3,
    const __bf16* __restrict__ ThsT, const __bf16* __restrict__ ThdT,
    float* __restrict__ out)
{
    const int M = 2048;
    const int lane = threadIdx.x & 31;
    const int wave = threadIdx.x >> 5;
    const int flat = blockIdx.x * WPB + wave;    // 0..511
    const int mtile = flat >> 2;
    const int nt    = flat & 3;
    const int m0    = mtile * 16;
    const int b     = blockIdx.y;
    const int mrow  = lane & 15;
    const int half  = lane >> 4;

    const long long xoff = (long long)b * M * 64;
    const __bf16* As[4] = { Y0 + xoff, Y1 + xoff, Y2 + xoff, Y3 + xoff };
    const __bf16* Ad[4] = { Y0 + xoff, Z1 + xoff, Z2 + xoff, Z3 + xoff };

    v8f accS = {};
    v8f accD = {};
    #pragma unroll
    for (int k = 0; k < 4; ++k) {
        const __bf16* aS = As[k] + (long long)(m0 + mrow) * 64;
        const __bf16* aD = Ad[k] + (long long)(m0 + mrow) * 64;
        const __bf16* bs = ThsT + k * 4096 + (nt * 16 + mrow) * 64;
        const __bf16* bd = ThdT + k * 4096 + (nt * 16 + mrow) * 64;
        #pragma unroll
        for (int kb = 0; kb < 64; kb += 32) {
            v16bf afS = load_afrag(aS, kb, half);
            v16bf bS  = *(const v16bf*)(bs + kb + 16 * half);
            accS = wmma_bf16(afS, bS, accS);
            v16bf afD = load_afrag(aD, kb, half);
            v16bf bD  = *(const v16bf*)(bd + kb + 16 * half);
            accD = wmma_bf16(afD, bD, accD);
        }
    }

    const int mbase = m0 + 8 * half;
    const int n = nt * 16 + mrow;
    #pragma unroll
    for (int r = 0; r < 8; ++r) {
        float v = fmaxf(accS[r], 0.0f) + fmaxf(accD[r], 0.0f);
        out[xoff + (long long)(mbase + r) * 64 + n] = v;
    }
}

// ---- launch ----------------------------------------------------------------

extern "C" void kernel_launch(void* const* d_in, const int* in_sizes, int n_in,
                              void* d_out, int out_size, void* d_ws, size_t ws_size,
                              hipStream_t stream) {
    (void)in_sizes; (void)n_in; (void)out_size; (void)ws_size;
    const float* X   = (const float*)d_in[0];   // [8,2048,64]
    const float* Ws  = (const float*)d_in[1];   // [2048,2048]
    const float* Wa  = (const float*)d_in[2];   // [2048,2048]
    const float* Ths = (const float*)d_in[3];   // [4,64,64]
    const float* Thd = (const float*)d_in[4];   // [4,64,64]
    float* out = (float*)d_out;                 // [8,2048,64]

    const int NN = 2048, C = 64, B = 8, KCH = 4;
    const long long NC  = (long long)NN * C;    // per-batch elems
    const long long NCB = NC * B;
    const long long CC  = (long long)C * C;

    char* ws = (char*)d_ws;
    size_t off = 0;
    auto alloc = [&](size_t bytes) -> char* {
        char* p = ws + off;
        off = (off + bytes + 255) & ~(size_t)255;
        return p;
    };
    __bf16* Wsb  = (__bf16*)alloc((size_t)NN * NN * 2);
    __bf16* Wab  = (__bf16*)alloc((size_t)NN * NN * 2);
    __bf16* Xb   = (__bf16*)alloc(NCB * 2);   // X row-major bf16
    __bf16* Xt   = (__bf16*)alloc(NCB * 2);   // X^T  [B,64,2048]
    __bf16* WaXt = (__bf16*)alloc(NCB * 2);   // (W_alpha X)^T [B,64,2048]
    float*  Y1f  = (float*) alloc(NCB * 4);
    float*  Z1f  = (float*) alloc(NCB * 4);
    __bf16* Y1rm = (__bf16*)alloc(NCB * 2);
    __bf16* Y2rm = (__bf16*)alloc(NCB * 2);
    __bf16* Y3rm = (__bf16*)alloc(NCB * 2);
    __bf16* Z1rm = (__bf16*)alloc(NCB * 2);
    __bf16* Z2rm = (__bf16*)alloc(NCB * 2);
    __bf16* Z3rm = (__bf16*)alloc(NCB * 2);
    __bf16* Y1t  = (__bf16*)alloc(NCB * 2);
    __bf16* Y2t  = (__bf16*)alloc(NCB * 2);
    __bf16* Z1t  = (__bf16*)alloc(NCB * 2);
    __bf16* Z2t  = (__bf16*)alloc(NCB * 2);
    __bf16* S0t  = (__bf16*)alloc((size_t)B * CC * 2);
    __bf16* S1t  = (__bf16*)alloc((size_t)B * CC * 2);
    __bf16* S2t  = (__bf16*)alloc((size_t)B * CC * 2);
    __bf16* ThsT = (__bf16*)alloc((size_t)KCH * CC * 2);
    __bf16* ThdT = (__bf16*)alloc((size_t)KCH * CC * 2);

    dim3 blk(256);
    const long long nWs = (long long)NN * NN;
    cvt_f32_bf16<<<dim3((unsigned)((nWs + 255) / 256)), blk, 0, stream>>>(Ws, Wsb, nWs);
    cvt_f32_bf16<<<dim3((unsigned)((nWs + 255) / 256)), blk, 0, stream>>>(Wa, Wab, nWs);
    cvt_f32_bf16<<<dim3((unsigned)((NCB + 255) / 256)), blk, 0, stream>>>(X, Xb, NCB);
    transpose_cvt<<<dim3((NN * C + 255) / 256, B),   blk, 0, stream>>>(X, Xt, NN, C);
    transpose_cvt<<<dim3((C * C + 255) / 256, KCH),  blk, 0, stream>>>(Ths, ThsT, C, C);
    transpose_cvt<<<dim3((C * C + 255) / 256, KCH),  blk, 0, stream>>>(Thd, ThdT, C, C);

    dim3 gBig((NN + 16 * WPB - 1) / (16 * WPB), B);   // (16, 8)
    dim3 gS  ((C  + 16 * WPB - 1) / (16 * WPB), B);   // (1, 8)

    // ---- static Chebyshev chain: Y1 = Ws X ; Y2 = 2 Ws Y1 - X ; Y3 = 2 Ws Y2 - Y1
    gemm_wmma<<<gBig, blk, 0, stream>>>(Wsb, 0, NN, Xt,  NC, NN, nullptr, 0,
                                        Y1f, Y1rm, NC, Y1t, NC, NN, NN, 1.0f);
    gemm_wmma<<<gBig, blk, 0, stream>>>(Wsb, 0, NN, Y1t, NC, NN, X, NC,
                                        nullptr, Y2rm, NC, Y2t, NC, NN, NN, 2.0f);
    gemm_wmma<<<gBig, blk, 0, stream>>>(Wsb, 0, NN, Y2t, NC, NN, Y1f, NC,
                                        nullptr, Y3rm, NC, nullptr, 0, NN, NN, 2.0f);

    // ---- dynamic chain: WaX = W_alpha X ; S_k = WaX^T Z_k ; Z_{k+1} = 2 X S_k - Z_{k-1}
    gemm_wmma<<<gBig, blk, 0, stream>>>(Wab, 0, NN, Xt,  NC, NN, nullptr, 0,
                                        nullptr, nullptr, 0, WaXt, NC, NN, NN, 1.0f);
    gemm_wmma<<<gS,   blk, 0, stream>>>(WaXt, NC, NN, Xt,  NC, NN, nullptr, 0,
                                        nullptr, nullptr, 0, S0t, CC, C, NN, 1.0f);
    gemm_wmma<<<gBig, blk, 0, stream>>>(Xb, NC, C, S0t, CC, C, nullptr, 0,
                                        Z1f, Z1rm, NC, Z1t, NC, NN, C, 1.0f);
    gemm_wmma<<<gS,   blk, 0, stream>>>(WaXt, NC, NN, Z1t, NC, NN, nullptr, 0,
                                        nullptr, nullptr, 0, S1t, CC, C, NN, 1.0f);
    gemm_wmma<<<gBig, blk, 0, stream>>>(Xb, NC, C, S1t, CC, C, X, NC,
                                        nullptr, Z2rm, NC, Z2t, NC, NN, C, 2.0f);
    gemm_wmma<<<gS,   blk, 0, stream>>>(WaXt, NC, NN, Z2t, NC, NN, nullptr, 0,
                                        nullptr, nullptr, 0, S2t, CC, C, NN, 1.0f);
    gemm_wmma<<<gBig, blk, 0, stream>>>(Xb, NC, C, S2t, CC, C, Z1f, NC,
                                        nullptr, Z3rm, NC, nullptr, 0, NN, C, 2.0f);

    // ---- fused theta contraction + relu + add
    final_wmma<<<dim3(64, B), blk, 0, stream>>>(Xb, Y1rm, Y2rm, Y3rm,
                                                Z1rm, Z2rm, Z3rm, ThsT, ThdT, out);
}
